// BilinearAttentionLayer_72834055406361
// MI455X (gfx1250) — matrix operand order repaired and checked by reading, not compiled, not run on hardware
//
#include <hip/hip_runtime.h>
#include <hip/hip_bf16.h>
#include <math.h>

// ---------------------------------------------------------------------------
// Bilinear attention, MI455X (gfx1250, wave32, WMMA).
//   out = softmax( (x @ W @ x^T) / sqrt(D) ) @ x
// B=8, S=2048, D=512, fp32 in/out. All GEMMs on v_wmma_f32_16x16x32_bf16
// (f32 accumulate). 1/sqrt(D) folded into xw. Softmax fused into the score
// GEMM epilogue. All WMMA operands are contiguous 2x16B loads (transposed
// bf16 copies staged in workspace). GEMM inner loops are software-pipelined
// with TWO named operand buffers indexed by compile-time-constant parity
// (no copies -> distinct registers -> partial s_wait_loadcnt, no v_mov
// rotation in the hot loop).
// Workspace layout (bytes):
//   [ 0M,16M)    x_bf   : bf16 x, row-major [B*S, D]
//   [16M,32M)    xw_bf  : bf16 (x@W)/sqrt(D), [B*S, D]
//   [32M,48M)    xt_bf  : bf16 x transposed  [B, D, S]
//   [48M,+.5M)   wt_bf  : bf16 W transposed  [D, D]  (wt[n][k] = W[k][n])
//   [~48.5M,+64M) P_bf  : bf16 softmax probs [B, S, S]
// ---------------------------------------------------------------------------

typedef __attribute__((ext_vector_type(16))) __bf16 bf16x16;
typedef __attribute__((ext_vector_type(8)))  __bf16 bf16x8;
typedef __attribute__((ext_vector_type(4)))  __bf16 bf16x4;
typedef __attribute__((ext_vector_type(8)))  float  f32x8;

static constexpr int BDIM = 8;
static constexpr int SDIM = 2048;
static constexpr int DDIM = 512;
static constexpr float SCALE = 0.04419417382415922f;  // 1/sqrt(512)

// ---------------------------------------------------------------------------
// Kernel 0: x (f32) -> x_bf (row-major bf16) AND xt_bf (transposed bf16).
// ---------------------------------------------------------------------------
__global__ __launch_bounds__(256) void cvt_transpose_x(
    const float* __restrict__ x, __bf16* __restrict__ xb,
    __bf16* __restrict__ xt) {
  __shared__ __bf16 tile[64][64 + 4];
  const int bidx = blockIdx.x;
  const int b  = bidx >> 8;
  const int s0 = ((bidx >> 3) & 31) * 64;
  const int d0 = (bidx & 7) * 64;

  const float* xbat = x  + (size_t)b * SDIM * DDIM;
  __bf16*      xbb  = xb + (size_t)b * SDIM * DDIM;
  __bf16*      xtb  = xt + (size_t)b * DDIM * SDIM;

  const int r  = threadIdx.x >> 2;
  const int c0 = (threadIdx.x & 3) * 16;

  {
    const float* src = xbat + (size_t)(s0 + r) * DDIM + d0 + c0;
    __bf16* dst = xbb + (size_t)(s0 + r) * DDIM + d0 + c0;
#pragma unroll
    for (int i = 0; i < 16; i += 4) {
      float4 v = *(const float4*)(src + i);
      bf16x4 o;
      o[0] = (__bf16)v.x; o[1] = (__bf16)v.y;
      o[2] = (__bf16)v.z; o[3] = (__bf16)v.w;
      *(bf16x4*)(dst + i) = o;
      tile[r][c0 + i + 0] = o[0];
      tile[r][c0 + i + 1] = o[1];
      tile[r][c0 + i + 2] = o[2];
      tile[r][c0 + i + 3] = o[3];
    }
  }
  __syncthreads();
  {
    __bf16* dst = xtb + (size_t)(d0 + r) * SDIM + s0 + c0;
#pragma unroll
    for (int i = 0; i < 16; i += 4) {
      bf16x4 o;
      o[0] = tile[c0 + i + 0][r];
      o[1] = tile[c0 + i + 1][r];
      o[2] = tile[c0 + i + 2][r];
      o[3] = tile[c0 + i + 3][r];
      *(bf16x4*)(dst + i) = o;
    }
  }
}

// ---------------------------------------------------------------------------
// Kernel 0b: W (f32, [D,D]) -> wt_bf (bf16 transposed).
// ---------------------------------------------------------------------------
__global__ __launch_bounds__(256) void cvt_transpose_w(
    const float* __restrict__ w, __bf16* __restrict__ wt) {
  __shared__ __bf16 tile[64][64 + 4];
  const int k0 = (blockIdx.x >> 3) * 64;
  const int n0 = (blockIdx.x & 7) * 64;
  const int r  = threadIdx.x >> 2;
  const int c0 = (threadIdx.x & 3) * 16;
  {
    const float* src = w + (size_t)(k0 + r) * DDIM + n0 + c0;
#pragma unroll
    for (int i = 0; i < 16; i += 4) {
      float4 v = *(const float4*)(src + i);
      tile[r][c0 + i + 0] = (__bf16)v.x;
      tile[r][c0 + i + 1] = (__bf16)v.y;
      tile[r][c0 + i + 2] = (__bf16)v.z;
      tile[r][c0 + i + 3] = (__bf16)v.w;
    }
  }
  __syncthreads();
  {
    __bf16* dst = wt + (size_t)(n0 + r) * DDIM + k0 + c0;
#pragma unroll
    for (int i = 0; i < 16; i += 4) {
      bf16x4 o;
      o[0] = tile[c0 + i + 0][r];
      o[1] = tile[c0 + i + 1][r];
      o[2] = tile[c0 + i + 2][r];
      o[3] = tile[c0 + i + 3][r];
      *(bf16x4*)(dst + i) = o;
    }
  }
}

// A fragment (16x32 bf16): lane l -> row, K chunk kb..kb+15 (kb = k0 + 16*hl).
__device__ __forceinline__ bf16x16 load_a_bf16(const __bf16* __restrict__ src,
                                               int ld, int row, int k0, int hl) {
  const __bf16* p = src + (size_t)row * ld + k0 + hl * 16;
  bf16x8 lo = *(const bf16x8*)p;
  bf16x8 hi = *(const bf16x8*)(p + 8);
  bf16x16 a;
#pragma unroll
  for (int i = 0; i < 8; ++i) { a[i] = lo[i]; a[8 + i] = hi[i]; }
  return a;
}

// B fragment (32x16 bf16) where B(k,n) = rows[n][k]; contiguous 32B per lane.
__device__ __forceinline__ bf16x16 load_b_from_rows(const __bf16* __restrict__ rows,
                                                    int ld, int col, int k0, int hl) {
  const __bf16* p = rows + (size_t)col * ld + k0 + hl * 16;
  bf16x8 lo = *(const bf16x8*)p;
  bf16x8 hi = *(const bf16x8*)(p + 8);
  bf16x16 b;
#pragma unroll
  for (int i = 0; i < 8; ++i) { b[i] = lo[i]; b[8 + i] = hi[i]; }
  return b;
}

// ---------------------------------------------------------------------------
// Kernel 1: xw_bf = (x_bf @ W) * (1/sqrt(D)); double-buffered pipeline.
// ---------------------------------------------------------------------------
__global__ __launch_bounds__(256) void gemm_xw(
    const __bf16* __restrict__ xb, const __bf16* __restrict__ wt,
    __bf16* __restrict__ xw) {
  const int lane = threadIdx.x & 31;
  const int wave = threadIdx.x >> 5;
  const int hl   = lane >> 4;
  const int m0   = blockIdx.x * 16;
  const int n0   = wave * 64;
  const int arow = m0 + (lane & 15);
  const int bcol = n0 + (lane & 15);
  constexpr int NK = DDIM / 32;
  constexpr int T  = 4;

  f32x8 acc[T];
#pragma unroll
  for (int t = 0; t < T; ++t)
#pragma unroll
    for (int i = 0; i < 8; ++i) acc[t][i] = 0.0f;

  bf16x16 aa[2];
  bf16x16 bb[2];
  aa[0] = load_a_bf16(xb, DDIM, arow, 0, hl);
  bb[0] = load_b_from_rows(wt, DDIM, bcol, 0, hl);

#pragma unroll 2
  for (int kt = 0; kt < NK; ++kt) {
    const int k0 = kt * 32;
    const int kn = (kt + 1 < NK) ? k0 + 32 : k0;   // clamped, branch-free
    aa[(kt + 1) & 1] = load_a_bf16(xb, DDIM, arow, kn, hl);
#pragma unroll
    for (int t = 0; t < T; ++t) {
      const int tn = (t + 1 < T) ? t + 1 : 0;
      const int kb = (t + 1 < T) ? k0 : kn;
      bb[(t + 1) & 1] = load_b_from_rows(wt, DDIM, bcol + tn * 16, kb, hl);
      acc[t] = __builtin_amdgcn_wmma_f32_16x16x32_bf16(
          false, aa[kt & 1], false, bb[t & 1], (short)0, acc[t], false, false);
    }
  }
#pragma unroll
  for (int t = 0; t < T; ++t)
#pragma unroll
    for (int r = 0; r < 8; ++r) {
      const int m = m0 + r + 8 * hl;
      const int n = n0 + t * 16 + (lane & 15);
      xw[(size_t)m * DDIM + n] = (__bf16)(acc[t][r] * SCALE);
    }
}

// ---------------------------------------------------------------------------
// Kernel 2: scores = xw @ x^T, fused row softmax, bf16 P out; pipelined.
// Block = 8 waves = 16 query rows x 2048 keys (wave owns 16x256).
// ---------------------------------------------------------------------------
__global__ __launch_bounds__(256) void attn_scores(
    const __bf16* __restrict__ xw, const __bf16* __restrict__ xb,
    __bf16* __restrict__ P) {
  const int lane = threadIdx.x & 31;
  const int wave = threadIdx.x >> 5;
  const int hl   = lane >> 4;
  const int b    = blockIdx.x >> 7;
  const int m0   = (blockIdx.x & 127) * 16;

  const __bf16* q    = xw + (size_t)b * SDIM * DDIM;
  const __bf16* kmat = xb + (size_t)b * SDIM * DDIM;
  __bf16*       Pb   = P  + (size_t)b * SDIM * SDIM;

  const int arow = m0 + (lane & 15);
  const int n0w  = wave * 256;
  const int bcol = n0w + (lane & 15);
  constexpr int NK = DDIM / 32;
  constexpr int T  = 16;

  f32x8 acc[T];
#pragma unroll
  for (int t = 0; t < T; ++t)
#pragma unroll
    for (int i = 0; i < 8; ++i) acc[t][i] = 0.0f;

  bf16x16 aa[2];
  bf16x16 bb[2];
  aa[0] = load_a_bf16(q, DDIM, arow, 0, hl);
  bb[0] = load_b_from_rows(kmat, DDIM, bcol, 0, hl);

#pragma unroll 2
  for (int kt = 0; kt < NK; ++kt) {
    const int k0 = kt * 32;
    const int kn = (kt + 1 < NK) ? k0 + 32 : k0;
    aa[(kt + 1) & 1] = load_a_bf16(q, DDIM, arow, kn, hl);
#pragma unroll
    for (int t = 0; t < T; ++t) {
      const int tn = (t + 1 < T) ? t + 1 : 0;
      const int kb = (t + 1 < T) ? k0 : kn;
      bb[(t + 1) & 1] = load_b_from_rows(kmat, DDIM, bcol + tn * 16, kb, hl);
      acc[t] = __builtin_amdgcn_wmma_f32_16x16x32_bf16(
          false, aa[kt & 1], false, bb[t & 1], (short)0, acc[t], false, false);
    }
  }

  // ---- fused softmax over the 16 x 2048 row-block ----
  __shared__ float wred[8][16];
  __shared__ float grow[16];

  float pmax[8];
#pragma unroll
  for (int r = 0; r < 8; ++r) {
    float m = acc[0][r];
#pragma unroll
    for (int t = 1; t < T; ++t) m = fmaxf(m, acc[t][r]);
    pmax[r] = m;
  }
#pragma unroll
  for (int s = 1; s < 16; s <<= 1)
#pragma unroll
    for (int r = 0; r < 8; ++r)
      pmax[r] = fmaxf(pmax[r], __shfl_xor(pmax[r], s, 32));

  if ((lane & 15) == 0) {
#pragma unroll
    for (int r = 0; r < 8; ++r) wred[wave][hl * 8 + r] = pmax[r];
  }
  __syncthreads();
  if (threadIdx.x < 16) {
    float g = wred[0][threadIdx.x];
#pragma unroll
    for (int w = 1; w < 8; ++w) g = fmaxf(g, wred[w][threadIdx.x]);
    grow[threadIdx.x] = g;
  }
  __syncthreads();

  float rmax[8], psum[8];
#pragma unroll
  for (int r = 0; r < 8; ++r) { rmax[r] = grow[hl * 8 + r]; psum[r] = 0.0f; }
#pragma unroll
  for (int t = 0; t < T; ++t)
#pragma unroll
    for (int r = 0; r < 8; ++r) {
      float e = __expf(acc[t][r] - rmax[r]);
      acc[t][r] = e;
      psum[r] += e;
    }
#pragma unroll
  for (int s = 1; s < 16; s <<= 1)
#pragma unroll
    for (int r = 0; r < 8; ++r) psum[r] += __shfl_xor(psum[r], s, 32);

  __syncthreads();
  if ((lane & 15) == 0) {
#pragma unroll
    for (int r = 0; r < 8; ++r) wred[wave][hl * 8 + r] = psum[r];
  }
  __syncthreads();
  if (threadIdx.x < 16) {
    float g = 0.0f;
#pragma unroll
    for (int w = 0; w < 8; ++w) g += wred[w][threadIdx.x];
    grow[threadIdx.x] = g;
  }
  __syncthreads();

  float rinv[8];
#pragma unroll
  for (int r = 0; r < 8; ++r) rinv[r] = 1.0f / grow[hl * 8 + r];

#pragma unroll
  for (int t = 0; t < T; ++t)
#pragma unroll
    for (int r = 0; r < 8; ++r) {
      const int m = m0 + r + 8 * hl;
      const int n = n0w + t * 16 + (lane & 15);
      Pb[(size_t)m * SDIM + n] = (__bf16)(acc[t][r] * rinv[r]);
    }
}

// ---------------------------------------------------------------------------
// Kernel 3: out = P @ x (f32 out). B(k,n) = xt[n][k]; pipelined.
// ---------------------------------------------------------------------------
__global__ __launch_bounds__(256) void attn_out(
    const __bf16* __restrict__ P, const __bf16* __restrict__ xt,
    float* __restrict__ out) {
  const int lane = threadIdx.x & 31;
  const int wave = threadIdx.x >> 5;
  const int hl   = lane >> 4;
  const int b    = blockIdx.x >> 7;
  const int m0   = (blockIdx.x & 127) * 16;

  const __bf16* Pb  = P  + (size_t)b * SDIM * SDIM;
  const __bf16* xtb = xt + (size_t)b * DDIM * SDIM;
  float*        ob  = out + (size_t)b * SDIM * DDIM;

  const int arow = m0 + (lane & 15);
  const int n0   = wave * 64;
  const int bcol = n0 + (lane & 15);
  constexpr int NK = SDIM / 32;
  constexpr int T  = 4;

  f32x8 acc[T];
#pragma unroll
  for (int t = 0; t < T; ++t)
#pragma unroll
    for (int i = 0; i < 8; ++i) acc[t][i] = 0.0f;

  bf16x16 aa[2];
  bf16x16 bb[2];
  aa[0] = load_a_bf16(Pb, SDIM, arow, 0, hl);
  bb[0] = load_b_from_rows(xtb, SDIM, bcol, 0, hl);

#pragma unroll 2
  for (int kt = 0; kt < NK; ++kt) {
    const int k0 = kt * 32;
    const int kn = (kt + 1 < NK) ? k0 + 32 : k0;
    aa[(kt + 1) & 1] = load_a_bf16(Pb, SDIM, arow, kn, hl);
#pragma unroll
    for (int t = 0; t < T; ++t) {
      const int tn = (t + 1 < T) ? t + 1 : 0;
      const int kb = (t + 1 < T) ? k0 : kn;
      bb[(t + 1) & 1] = load_b_from_rows(xtb, SDIM, bcol + tn * 16, kb, hl);
      acc[t] = __builtin_amdgcn_wmma_f32_16x16x32_bf16(
          false, aa[kt & 1], false, bb[t & 1], (short)0, acc[t], false, false);
    }
  }
#pragma unroll
  for (int t = 0; t < T; ++t)
#pragma unroll
    for (int r = 0; r < 8; ++r) {
      const int m = m0 + r + 8 * hl;
      const int n = n0 + t * 16 + (lane & 15);
      ob[(size_t)m * DDIM + n] = acc[t][r];
    }
}

// ---------------------------------------------------------------------------
extern "C" void kernel_launch(void* const* d_in, const int* in_sizes, int n_in,
                              void* d_out, int out_size, void* d_ws, size_t ws_size,
                              hipStream_t stream) {
  const float* x = (const float*)d_in[0];        // [B,S,D] fp32
  const float* w = (const float*)d_in[1];        // [D,D]   fp32
  float* out = (float*)d_out;                    // [B,S,D] fp32

  char* ws = (char*)d_ws;
  const size_t XBF = (size_t)BDIM * SDIM * DDIM * 2;   // 16 MiB
  const size_t WT  = (size_t)DDIM * DDIM * 2;          // 512 KiB
  __bf16* xb  = (__bf16*)(ws);
  __bf16* xw  = (__bf16*)(ws + XBF);
  __bf16* xt  = (__bf16*)(ws + 2 * XBF);
  __bf16* wt  = (__bf16*)(ws + 3 * XBF);
  __bf16* Pp  = (__bf16*)(ws + 3 * XBF + WT);

  cvt_transpose_x<<<BDIM * (SDIM / 64) * (DDIM / 64), 256, 0, stream>>>(x, xb, xt);
  cvt_transpose_w<<<(DDIM / 64) * (DDIM / 64), 256, 0, stream>>>(w, wt);
  gemm_xw<<<(BDIM * SDIM) / 16, 256, 0, stream>>>(xb, wt, xw);
  attn_scores<<<BDIM * (SDIM / 16), 256, 0, stream>>>(xw, xb, Pp);
  attn_out<<<BDIM * (SDIM / 16), 256, 0, stream>>>(Pp, xt, out);
}
